// LightweightTransformerChatbot_46265387712608
// MI455X (gfx1250) — matrix operand order repaired and checked by reading, not compiled, not run on hardware
//
#include <hip/hip_runtime.h>
#include <hip/hip_bf16.h>

typedef __attribute__((ext_vector_type(16))) _Float16 v16h;
typedef __attribute__((ext_vector_type(8)))  float    v8f;
typedef __attribute__((ext_vector_type(4)))  int      v4i;

namespace cfg {
constexpr int B = 4, S = 1024, D = 1024, H = 16, L = 4, F = 4096, V = 32000, DK = 64;
constexpr int M = B * S;  // 4096 token rows
}

// -------------------- gfx1250 async-LDS plumbing ---------------------------
#if __has_builtin(__builtin_amdgcn_global_load_async_to_lds_b128)
#define HAVE_ASYNC_LDS 1
typedef __attribute__((address_space(1))) v4i glob_v4i;  // global (device) int4
typedef __attribute__((address_space(3))) v4i lds_v4i;   // LDS int4
#else
#define HAVE_ASYNC_LDS 0
#endif

__device__ inline void async_wait0() {
#if HAVE_ASYNC_LDS
#if __has_builtin(__builtin_amdgcn_s_wait_asynccnt)
  __builtin_amdgcn_s_wait_asynccnt(0);
#else
  asm volatile("s_wait_asynccnt 0" ::: "memory");
#endif
#endif
}

// copy 16 bytes global -> LDS (async on gfx1250 if available)
__device__ inline void cp_g2l_16(const _Float16* __restrict__ g, _Float16* l) {
#if HAVE_ASYNC_LDS
  __builtin_amdgcn_global_load_async_to_lds_b128((glob_v4i*)g, (lds_v4i*)l, 0, 0);
#else
  *(int4*)l = *(const int4*)g;
#endif
}

// ---------------------------------------------------------------------------
// WMMA helpers (gfx1250, wave32). D = A(16x32 f16) * B(32x16 f16) + C(16x16 f32)
// ---------------------------------------------------------------------------
__device__ inline v8f wmma16(v16h a, v16h b, v8f c) {
  return __builtin_amdgcn_wmma_f32_16x16x32_f16(
      /*neg_a=*/false, a, /*neg_b=*/false, b,
      /*c_mod=*/(short)0, c, /*reuse_a=*/false, /*reuse_b=*/false);
}

// A operand from row-major [16 rows(M), ld] tile (16-bit A-matrix 16x32 layout):
// lane r = lane%16 = M row; g = lane/16; VGPR j -> K = g*16 + 2j{,+1}
// (contiguous, matching B). Pairs are contiguous -> dword loads.
__device__ inline v16h load_a_rm(const _Float16* __restrict__ base, int ld, int lane) {
  int r = lane & 15, g = (lane >> 4) & 1;
  const _Float16* p = base + (size_t)r * ld;
  v16h f;
#pragma unroll
  for (int j = 0; j < 4; ++j) {
    int k0 = g * 16 + 2 * j;
    f[2 * j]     = p[k0];
    f[2 * j + 1] = p[k0 + 1];
    int k1 = g * 16 + 8 + 2 * j;
    f[8 + 2 * j]     = p[k1];
    f[8 + 2 * j + 1] = p[k1 + 1];
  }
  return f;
}

// B operand from row-major [16 rows(N), ld] tile (i.e. B^T stored row-major).
// B in this code's contiguous k order (same as A): lanes 0-15 hold K=0-15, lanes 16-31 hold K=16-31;
// VGPR j holds K = g*16 + 2j{,+1}.  16 contiguous halfs per lane.
__device__ inline v16h load_b_nk(const _Float16* __restrict__ base, int ld, int lane) {
  int r = lane & 15, g = (lane >> 4) & 1;
  const _Float16* p = base + (size_t)r * ld + g * 16;
  v16h f;
#pragma unroll
  for (int j = 0; j < 8; ++j) {
    f[2 * j]     = p[2 * j];
    f[2 * j + 1] = p[2 * j + 1];
  }
  return f;
}

// B operand from row-major [32 rows(K), ld] tile (K-major source, e.g. V tiles).
__device__ inline v16h load_b_kn(const _Float16* __restrict__ base, int ld, int lane) {
  int n = lane & 15, g = (lane >> 4) & 1;
  v16h f;
#pragma unroll
  for (int j = 0; j < 8; ++j) {
    int kk = g * 16 + 2 * j;
    f[2 * j]     = base[(size_t)kk * ld + n];
    f[2 * j + 1] = base[(size_t)(kk + 1) * ld + n];
  }
  return f;
}

// ---------------------------------------------------------------------------
// Weight transpose + fp32 -> f16 cast:  Wt[n*K + k] = W[k*N + n]
// ---------------------------------------------------------------------------
__global__ void k_transpose_cast(const float* __restrict__ W, _Float16* __restrict__ Wt,
                                 int K, int N) {
  size_t total = (size_t)K * N;
  size_t idx = (size_t)blockIdx.x * blockDim.x + threadIdx.x;
  if (idx >= total) return;
  int k = (int)(idx / N);
  int n = (int)(idx % N);
  Wt[(size_t)n * K + k] = (_Float16)W[idx];
}

// ---------------------------------------------------------------------------
// Embedding * sqrt(D) + sinusoidal positional encoding
// ---------------------------------------------------------------------------
__global__ void k_embed(const int* __restrict__ x, const float* __restrict__ emb,
                        float* __restrict__ h, _Float16* __restrict__ h16) {
  using namespace cfg;
  int row = blockIdx.x;          // token row in [0, B*S)
  int s = row % S;
  int tok = x[row];
  const float inv = -9.2103403719761836f / (float)D;  // -ln(10000)/D
  for (int d = threadIdx.x; d < D; d += blockDim.x) {
    float div = expf((float)(d & ~1) * inv);
    float ang = (float)s * div;
    float pe = (d & 1) ? cosf(ang) : sinf(ang);
    float val = emb[(size_t)tok * D + d] * 32.0f + pe;  // sqrt(1024) = 32
    h[(size_t)row * D + d] = val;
    h16[(size_t)row * D + d] = (_Float16)val;
  }
}

// ---------------------------------------------------------------------------
// WMMA GEMM with LDS-staged, double-buffered B tiles.
// C[M,N] = A[M,K](f16) * Bt[N,K]^T(f16) (+bias) (+GELU)
// Block: 256 thr = 8 waves. Block tile 128M x 64N. Each wave: 16M x 64N
// (one A frag reused across 4 WMMAs). B tile (64N x 32K = 4KB) staged in LDS
// via async global->LDS loads, shared by all 8 waves. grid = (N/64, M/128)
// ---------------------------------------------------------------------------
__global__ __launch_bounds__(256) void k_gemm(const _Float16* __restrict__ A,
                                              const _Float16* __restrict__ Bt,
                                              const float* __restrict__ bias,
                                              float* __restrict__ C32,
                                              _Float16* __restrict__ C16,
                                              int M, int N, int K, int act) {
  __shared__ _Float16 Bl[2][64 * 32];
  int t = threadIdx.x;
  int lane = t & 31;
  int wave = t >> 5;
  int tn = blockIdx.x * 64;
  int tm = blockIdx.y * 128 + wave * 16;

  // staging coords: thread t copies 16B: B row r (N dim), k-chunk c
  int sr = t >> 2;            // 0..63
  int sc = (t & 3) * 8;       // f16 offset in 32-wide k slice
  const _Float16* bsrc = Bt + (size_t)(tn + sr) * K + sc;
  _Float16* bdst0 = &Bl[0][sr * 32 + sc];
  _Float16* bdst1 = &Bl[1][sr * 32 + sc];

  const _Float16* a0 = A + (size_t)tm * K;
  v8f acc0 = {}, acc1 = {}, acc2 = {}, acc3 = {};

  cp_g2l_16(bsrc, bdst0);  // stage first tile
  int buf = 0;
  for (int k0 = 0; k0 < K; k0 += 32) {
    async_wait0();       // my async copies into current buffer are done
    __syncthreads();     // everyone's copies visible
    if (k0 + 32 < K) {   // prefetch next B tile into other buffer
      cp_g2l_16(bsrc + (k0 + 32), buf ? bdst0 : bdst1);
      __builtin_prefetch(a0 + k0 + 32, 0, 0);  // global_prefetch for A stream
    }
    v16h a = load_a_rm(a0 + k0, K, lane);
    const _Float16* bl = Bl[buf];
    acc0 = wmma16(a, load_b_nk(bl +  0 * 32, 32, lane), acc0);
    acc1 = wmma16(a, load_b_nk(bl + 16 * 32, 32, lane), acc1);
    acc2 = wmma16(a, load_b_nk(bl + 32 * 32, 32, lane), acc2);
    acc3 = wmma16(a, load_b_nk(bl + 48 * 32, 32, lane), acc3);
    __syncthreads();     // all waves done reading before buffer is reused
    buf ^= 1;
  }

  // C layout: lane%16 = N col; row = 8*(lane/16) + vgpr
  int n = lane & 15;
  int mbase = (lane >> 4) * 8;
  v8f accs[4] = {acc0, acc1, acc2, acc3};
#pragma unroll
  for (int f = 0; f < 4; ++f) {
    float bv = bias ? bias[tn + f * 16 + n] : 0.0f;
#pragma unroll
    for (int vi = 0; vi < 8; ++vi) {
      float xv = accs[f][vi] + bv;
      if (act) xv = 0.5f * xv * (1.0f + erff(xv * 0.70710678118f));  // exact GELU
      size_t o = (size_t)(tm + mbase + vi) * N + tn + f * 16 + n;
      if (C32) C32[o] = xv;
      if (C16) C16[o] = (_Float16)xv;
    }
  }
}

// ---------------------------------------------------------------------------
// Flash-style masked attention. One wave per (b, h, 16-query tile).
// q/k/v are f16 [B*S, D]; head h occupies columns [h*DK, h*DK+64).
// ---------------------------------------------------------------------------
__global__ __launch_bounds__(32) void k_attn(const _Float16* __restrict__ q,
                                             const _Float16* __restrict__ k,
                                             const _Float16* __restrict__ v,
                                             const int* __restrict__ x,
                                             _Float16* __restrict__ ctx) {
  using namespace cfg;
  int bh = blockIdx.x;
  int b = bh >> 4;          // H == 16
  int h = bh & 15;
  int qbase = blockIdx.y * 16;
  int lane = threadIdx.x;
  int n = lane & 15;
  int mbase = (lane >> 4) * 8;

  const size_t qrow = (size_t)(b * S + qbase) * D + h * DK;
  v16h Aq0 = load_a_rm(q + qrow, D, lane);       // K = dk 0..31
  v16h Aq1 = load_a_rm(q + qrow + 32, D, lane);  // K = dk 32..63

  float m_i[8], l_i[8];
  v8f O[4] = {};
#pragma unroll
  for (int vi = 0; vi < 8; ++vi) { m_i[vi] = -3.0e38f; l_i[vi] = 0.0f; }

  __shared__ _Float16 Pl[16 * 32];  // P tile, row-major [16 q, 32 keys]

  for (int kb = 0; kb < qbase + 16; kb += 32) {
    // scores for two 16-key tiles: S = Q * K^T over dk=64
    const _Float16* K0 = k + (size_t)(b * S + kb) * D + h * DK;
    const _Float16* K1 = K0 + (size_t)16 * D;
    v8f c0 = {}, c1 = {};
    c0 = wmma16(Aq0, load_b_nk(K0, D, lane), c0);
    c0 = wmma16(Aq1, load_b_nk(K0 + 32, D, lane), c0);
    c1 = wmma16(Aq0, load_b_nk(K1, D, lane), c1);
    c1 = wmma16(Aq1, load_b_nk(K1 + 32, D, lane), c1);

    int key0 = kb + n, key1 = kb + 16 + n;
    bool pad0 = (x[b * S + key0] != 0);
    bool pad1 = (x[b * S + key1] != 0);

#pragma unroll
    for (int vi = 0; vi < 8; ++vi) {
      int qi = qbase + mbase + vi;
      float s0 = (pad0 && key0 <= qi) ? c0[vi] * 0.125f : -3.0e38f;  // /sqrt(64)
      float s1 = (pad1 && key1 <= qi) ? c1[vi] * 0.125f : -3.0e38f;
      // row max over 32 keys (row lives in one 16-lane half)
      float tt = fmaxf(s0, s1);
      tt = fmaxf(tt, __shfl_xor(tt, 1, 32));
      tt = fmaxf(tt, __shfl_xor(tt, 2, 32));
      tt = fmaxf(tt, __shfl_xor(tt, 4, 32));
      tt = fmaxf(tt, __shfl_xor(tt, 8, 32));
      float mn = fmaxf(m_i[vi], tt);
      float alpha = expf(m_i[vi] - mn);
      float p0 = expf(s0 - mn);
      float p1 = expf(s1 - mn);
      float rs = p0 + p1;
      rs += __shfl_xor(rs, 1, 32);
      rs += __shfl_xor(rs, 2, 32);
      rs += __shfl_xor(rs, 4, 32);
      rs += __shfl_xor(rs, 8, 32);
      l_i[vi] = l_i[vi] * alpha + rs;
      m_i[vi] = mn;
      Pl[(mbase + vi) * 32 + n] = (_Float16)p0;
      Pl[(mbase + vi) * 32 + 16 + n] = (_Float16)p1;
      O[0][vi] *= alpha; O[1][vi] *= alpha; O[2][vi] *= alpha; O[3][vi] *= alpha;
    }
    __syncthreads();

    // O += P(16x32) * V(32 keys x 64 dk)
    v16h Ap = load_a_rm(Pl, 32, lane);
    const _Float16* Vb = v + (size_t)(b * S + kb) * D + h * DK;
#pragma unroll
    for (int f = 0; f < 4; ++f)
      O[f] = wmma16(Ap, load_b_kn(Vb + f * 16, D, lane), O[f]);
    __syncthreads();
  }

#pragma unroll
  for (int f = 0; f < 4; ++f) {
#pragma unroll
    for (int vi = 0; vi < 8; ++vi) {
      float val = (l_i[vi] > 0.0f) ? O[f][vi] / l_i[vi] : 0.0f;
      ctx[(size_t)(b * S + qbase + mbase + vi) * D + h * DK + f * 16 + n] =
          (_Float16)val;
    }
  }
}

// ---------------------------------------------------------------------------
// Residual add + LayerNorm; writes fp32 residual stream + f16 GEMM operand
// ---------------------------------------------------------------------------
__global__ __launch_bounds__(256) void k_add_ln(float* __restrict__ h,
                                                const float* __restrict__ r,
                                                const float* __restrict__ gamma,
                                                const float* __restrict__ beta,
                                                _Float16* __restrict__ h16) {
  using namespace cfg;
  int row = blockIdx.x;
  int tid = threadIdx.x;
  float vals[4];
  float s = 0.0f;
#pragma unroll
  for (int i = 0; i < 4; ++i) {
    int d = tid + i * 256;
    float t = h[(size_t)row * D + d] + r[(size_t)row * D + d];
    vals[i] = t;
    s += t;
  }
  __shared__ float red[8];
  for (int off = 16; off > 0; off >>= 1) s += __shfl_xor(s, off, 32);
  if ((tid & 31) == 0) red[tid >> 5] = s;
  __syncthreads();
  float tot = 0.0f;
#pragma unroll
  for (int i = 0; i < 8; ++i) tot += red[i];
  float mu = tot * (1.0f / D);
  __syncthreads();
  float sq = 0.0f;
#pragma unroll
  for (int i = 0; i < 4; ++i) { float d0 = vals[i] - mu; sq += d0 * d0; }
  for (int off = 16; off > 0; off >>= 1) sq += __shfl_xor(sq, off, 32);
  if ((tid & 31) == 0) red[tid >> 5] = sq;
  __syncthreads();
  tot = 0.0f;
#pragma unroll
  for (int i = 0; i < 8; ++i) tot += red[i];
  float rstd = rsqrtf(tot * (1.0f / D) + 1e-5f);
#pragma unroll
  for (int i = 0; i < 4; ++i) {
    int d = tid + i * 256;
    float y = (vals[i] - mu) * rstd * gamma[d] + beta[d];
    h[(size_t)row * D + d] = y;
    h16[(size_t)row * D + d] = (_Float16)y;
  }
}

// ---------------------------------------------------------------------------
// Launcher
// ---------------------------------------------------------------------------
extern "C" void kernel_launch(void* const* d_in, const int* in_sizes, int n_in,
                              void* d_out, int out_size, void* d_ws, size_t ws_size,
                              hipStream_t stream) {
  (void)in_sizes; (void)n_in; (void)out_size; (void)ws_size;
  using namespace cfg;

  const int*   x     = (const int*)d_in[0];
  const float* emb   = (const float*)d_in[1];
  const float* Wq    = (const float*)d_in[2];
  const float* Wk    = (const float*)d_in[3];
  const float* Wv    = (const float*)d_in[4];
  const float* Wo    = (const float*)d_in[5];
  const float* bo    = (const float*)d_in[6];
  const float* ln1g  = (const float*)d_in[7];
  const float* ln1b  = (const float*)d_in[8];
  const float* ln2g  = (const float*)d_in[9];
  const float* ln2b  = (const float*)d_in[10];
  const float* W1    = (const float*)d_in[11];
  const float* b1    = (const float*)d_in[12];
  const float* W2    = (const float*)d_in[13];
  const float* b2    = (const float*)d_in[14];
  const float* Wout  = (const float*)d_in[15];
  const float* bout  = (const float*)d_in[16];

  char* ws = (char*)d_ws;
  auto alloc = [&](size_t bytes) -> void* {
    void* p = (void*)ws;
    ws += (bytes + 255) & ~(size_t)255;
    return p;
  };

  float*    hf32  = (float*)alloc((size_t)M * D * 4);
  _Float16* hf16  = (_Float16*)alloc((size_t)M * D * 2);
  _Float16* qf16  = (_Float16*)alloc((size_t)M * D * 2);
  _Float16* kf16  = (_Float16*)alloc((size_t)M * D * 2);
  _Float16* vf16  = (_Float16*)alloc((size_t)M * D * 2);
  _Float16* cf16  = (_Float16*)alloc((size_t)M * D * 2);
  float*    tmp32 = (float*)alloc((size_t)M * D * 4);
  _Float16* ff16  = (_Float16*)alloc((size_t)M * F * 2);
  _Float16* WqT   = (_Float16*)alloc((size_t)L * D * D * 2);
  _Float16* WkT   = (_Float16*)alloc((size_t)L * D * D * 2);
  _Float16* WvT   = (_Float16*)alloc((size_t)L * D * D * 2);
  _Float16* WoT   = (_Float16*)alloc((size_t)L * D * D * 2);
  _Float16* W1T   = (_Float16*)alloc((size_t)L * D * F * 2);
  _Float16* W2T   = (_Float16*)alloc((size_t)L * F * D * 2);
  _Float16* WouT  = (_Float16*)alloc((size_t)D * V * 2);

  auto tgrid = [](size_t total) { return dim3((unsigned)((total + 255) / 256)); };

  // weight prep: transpose + cast
  for (int l = 0; l < L; ++l) {
    size_t o = (size_t)l * D * D;
    k_transpose_cast<<<tgrid((size_t)D * D), 256, 0, stream>>>(Wq + o, WqT + o, D, D);
    k_transpose_cast<<<tgrid((size_t)D * D), 256, 0, stream>>>(Wk + o, WkT + o, D, D);
    k_transpose_cast<<<tgrid((size_t)D * D), 256, 0, stream>>>(Wv + o, WvT + o, D, D);
    k_transpose_cast<<<tgrid((size_t)D * D), 256, 0, stream>>>(Wo + o, WoT + o, D, D);
    size_t of = (size_t)l * D * F;
    k_transpose_cast<<<tgrid((size_t)D * F), 256, 0, stream>>>(W1 + of, W1T + of, D, F);
    k_transpose_cast<<<tgrid((size_t)F * D), 256, 0, stream>>>(W2 + of, W2T + of, F, D);
  }
  k_transpose_cast<<<tgrid((size_t)D * V), 256, 0, stream>>>(Wout, WouT, D, V);

  // embedding + positional encoding
  k_embed<<<M, 256, 0, stream>>>(x, emb, hf32, hf16);

  dim3 gDD(D / 64, M / 128);       // GEMM -> [M, D]
  dim3 gDF(F / 64, M / 128);       // GEMM -> [M, F]
  dim3 gDV(V / 64, M / 128);       // GEMM -> [M, V]
  dim3 gAT(B * H, S / 16);         // attention

  for (int l = 0; l < L; ++l) {
    size_t o = (size_t)l * D * D;
    size_t of = (size_t)l * D * F;
    // QKV projections (f16 outputs only)
    k_gemm<<<gDD, 256, 0, stream>>>(hf16, WqT + o, nullptr, nullptr, qf16, M, D, D, 0);
    k_gemm<<<gDD, 256, 0, stream>>>(hf16, WkT + o, nullptr, nullptr, kf16, M, D, D, 0);
    k_gemm<<<gDD, 256, 0, stream>>>(hf16, WvT + o, nullptr, nullptr, vf16, M, D, D, 0);
    // masked multi-head attention
    k_attn<<<gAT, 32, 0, stream>>>(qf16, kf16, vf16, x, cf16);
    // output projection + bias (fp32 for residual)
    k_gemm<<<gDD, 256, 0, stream>>>(cf16, WoT + o, bo + (size_t)l * D, tmp32, nullptr, M, D, D, 0);
    // h = LN(h + attn_out)
    k_add_ln<<<M, 256, 0, stream>>>(hf32, tmp32, ln1g + (size_t)l * D, ln1b + (size_t)l * D, hf16);
    // FFN: GELU(h @ W1 + b1) @ W2 + b2
    k_gemm<<<gDF, 256, 0, stream>>>(hf16, W1T + of, b1 + (size_t)l * F, nullptr, ff16, M, F, D, 1);
    k_gemm<<<gDD, 256, 0, stream>>>(ff16, W2T + of, b2 + (size_t)l * D, tmp32, nullptr, M, D, F, 0);
    // h = LN(h + ff)
    k_add_ln<<<M, 256, 0, stream>>>(hf32, tmp32, ln2g + (size_t)l * D, ln2b + (size_t)l * D, hf16);
  }

  // logits = h @ Wout + bout  -> d_out (fp32 [M, V])
  k_gemm<<<gDV, 256, 0, stream>>>(hf16, WouT, bout, (float*)d_out, nullptr, M, V, D, 0);
}
